// GraphExportAdapter_6090263626040
// MI455X (gfx1250) — matrix operand order repaired and checked by reading, not compiled, not run on hardware
//
#include <hip/hip_runtime.h>
#include <hip/hip_bf16.h>
#include <math.h>

typedef __attribute__((ext_vector_type(16))) __bf16 v16bf;
typedef __attribute__((ext_vector_type(8)))  float  v8f;
typedef unsigned short u16b;
typedef unsigned int   u32b;

// ---------- bf16 helpers (round-to-nearest-even), storage as ushort ----------
__device__ __forceinline__ u16b f2bf(float f) {
  u32b u = __float_as_uint(f);
  u32b r = (u + 0x7FFFu + ((u >> 16) & 1u)) >> 16;
  return (u16b)r;
}
__device__ __forceinline__ float bf2f(u16b s) {
  return __uint_as_float(((u32b)s) << 16);
}

union FragAB {
  v16bf v;
  uint4 u[2];
  u16b  s[16];
};
union FragC {
  v8f   v;
  float f[8];
};

// A fragment: 16x32 bf16 tile, row-major source with leading dim lda (elements).
// Lane l: m = l&15 ; halves 0..7  -> K = kk..kk+7   (kk = (l>=16)?8:0)
//                    halves 8..15 -> K = 16+kk..16+kk+7
__device__ __forceinline__ void load_fragA(FragAB& fa, const u16b* a, int lda, int lane) {
  const u16b* p = a + (size_t)(lane & 15) * lda + ((lane >> 4) << 3);
  fa.u[0] = *reinterpret_cast<const uint4*>(p);
  fa.u[1] = *reinterpret_cast<const uint4*>(p + 16);
}

// B fragment: 32x16 bf16 tile; source is B^T stored [N,K] row-major (ldk).
// Lane l: n = l&15 ; halves 0..7  -> K = kk..kk+7   (kk = (l>=16)?8:0)
//                    halves 8..15 -> K = 16+kk..16+kk+7
__device__ __forceinline__ void load_fragB(FragAB& fb, const u16b* bt, int ldk, int lane) {
  const u16b* p = bt + (size_t)(lane & 15) * ldk + ((lane >> 4) << 3);
  fb.u[0] = *reinterpret_cast<const uint4*>(p);
  fb.u[1] = *reinterpret_cast<const uint4*>(p + 16);
}

__device__ __forceinline__ v8f do_wmma(const FragAB& a, const FragAB& b, v8f c) {
  return __builtin_amdgcn_wmma_f32_16x16x32_bf16(false, a.v, false, b.v,
                                                 (short)0, c, false, false);
}

// ---------- transpose + f32 -> bf16 weight conversion ----------
__global__ void convT_kernel(const float* __restrict__ W, u16b* __restrict__ WT,
                             int K, int N) {
  int idx = blockIdx.x * blockDim.x + threadIdx.x;
  if (idx >= K * N) return;
  int k = idx / N;
  int n = idx - k * N;
  WT[(size_t)n * K + k] = f2bf(W[idx]);
}

__global__ void copy_f32_kernel(const float* __restrict__ src, float* __restrict__ dst, int n) {
  int i = blockIdx.x * blockDim.x + threadIdx.x;
  if (i < n) dst[i] = src[i];
}

// ---------- LayerNorm: f32 in -> bf16 out, C=512, one wave per row ----------
__global__ __launch_bounds__(256) void ln_kernel(const float* __restrict__ x,
                                                 const float* __restrict__ scale,
                                                 const float* __restrict__ bias,
                                                 u16b* __restrict__ out) {
  int row  = blockIdx.x * 8 + (threadIdx.x >> 5);
  int lane = threadIdx.x & 31;
  const float* xr = x + (size_t)row * 512;
  float xv[16];
  float s = 0.f;
#pragma unroll
  for (int j = 0; j < 16; ++j) { xv[j] = xr[lane + j * 32]; s += xv[j]; }
#pragma unroll
  for (int o = 1; o < 32; o <<= 1) s += __shfl_xor(s, o, 32);
  float mu = s * (1.0f / 512.0f);
  float vv = 0.f;
#pragma unroll
  for (int j = 0; j < 16; ++j) { float d = xv[j] - mu; vv += d * d; }
#pragma unroll
  for (int o = 1; o < 32; o <<= 1) vv += __shfl_xor(vv, o, 32);
  float rs = rsqrtf(vv * (1.0f / 512.0f) + 1e-5f);
  u16b* orow = out + (size_t)row * 512;
#pragma unroll
  for (int j = 0; j < 16; ++j) {
    int c = lane + j * 32;
    orow[c] = f2bf((xv[j] - mu) * rs * scale[c] + bias[c]);
  }
}

// ---------- Generic WMMA GEMM: out = epi(A[M,K] @ WT[N,K]^T + bias) ----------
// 256 threads = 8 waves (4 Mwaves x 2 Nwaves), wave tile 32Mx64N (8 C tiles),
// block tile 128Mx128N, K step 32. Each B fragment feeds 2 WMMAs.
// EPI 0: bf16 out             EPI 1: QKV scatter (q scaled, V transposed)
// EPI 2: +residual -> f32     EPI 3: gelu -> bf16
// EPI 4: (+residual)*mask -> f32
template <int EPI>
__global__ __launch_bounds__(256) void gemm_bf16(
    const u16b* __restrict__ A, const u16b* __restrict__ WT,
    const float* __restrict__ bias, int M, int N, int K,
    u16b* __restrict__ out_bf, float* __restrict__ out_f32,
    const float* __restrict__ residual, const int* __restrict__ mask,
    u16b* __restrict__ qo, u16b* __restrict__ ko, u16b* __restrict__ vo,
    float qscale) {
  int tid  = threadIdx.x;
  int lane = tid & 31;
  int wid  = tid >> 5;
  int m0 = blockIdx.y * 128 + (wid >> 1) * 32;   // wave covers rows m0..m0+31
  int n0 = blockIdx.x * 128 + (wid & 1) * 64;    // wave covers cols n0..n0+63

  FragC c[2][4];
#pragma unroll
  for (int r = 0; r < 2; ++r)
#pragma unroll
    for (int t = 0; t < 4; ++t)
#pragma unroll
      for (int j = 0; j < 8; ++j) c[r][t].f[j] = 0.f;

  for (int k0 = 0; k0 < K; k0 += 32) {
    // prefetch next k-step's lines into cache hierarchy
    if (k0 + 32 < K) {
      __builtin_prefetch(A + (size_t)m0 * K + k0 + 32, 0, 1);
      __builtin_prefetch(WT + (size_t)n0 * K + k0 + 32, 0, 1);
    }
    FragAB fa0, fa1;
    load_fragA(fa0, A + (size_t)m0 * K + k0, K, lane);
    load_fragA(fa1, A + (size_t)(m0 + 16) * K + k0, K, lane);
#pragma unroll
    for (int t = 0; t < 4; ++t) {
      FragAB fb;
      load_fragB(fb, WT + (size_t)(n0 + t * 16) * K + k0, K, lane);
      c[0][t].v = do_wmma(fa0, fb, c[0][t].v);
      c[1][t].v = do_wmma(fa1, fb, c[1][t].v);
    }
  }

#pragma unroll
  for (int r = 0; r < 2; ++r) {
#pragma unroll
    for (int t = 0; t < 4; ++t) {
#pragma unroll
      for (int j = 0; j < 8; ++j) {
        int m = m0 + r * 16 + j + ((lane >> 4) << 3);
        int n = n0 + t * 16 + (lane & 15);
        float val = c[r][t].f[j] + bias[n];
        if (EPI == 0) {
          out_bf[(size_t)m * N + n] = f2bf(val);
        } else if (EPI == 1) {
          int part = n >> 9;  // 0=q 1=k 2=v
          int w = n & 511;
          int h = w >> 6, d = w & 63;
          int b = m >> 10, l = m & 1023;
          size_t bh = (size_t)(b * 8 + h);
          if (part == 0)      qo[(bh * 1024 + l) * 64 + d] = f2bf(val * qscale);
          else if (part == 1) ko[(bh * 1024 + l) * 64 + d] = f2bf(val);
          else                vo[(bh * 64 + d) * 1024 + l] = f2bf(val);
        } else if (EPI == 2) {
          out_f32[(size_t)m * N + n] = residual[(size_t)m * N + n] + val;
        } else if (EPI == 3) {
          float x = val;
          float g = 0.5f * x *
                    (1.0f + tanhf(0.7978845608028654f * (x + 0.044715f * x * x * x)));
          out_bf[(size_t)m * N + n] = f2bf(g);
        } else {  // EPI == 4
          float rr = residual[(size_t)m * N + n] + val;
          rr *= (mask[m] != 0) ? 1.0f : 0.0f;
          out_f32[(size_t)m * N + n] = rr;
        }
      }
    }
  }
}

// ---------- Fused attention: one block = 16 queries of one (b,h) ----------
__global__ __launch_bounds__(128) void attn_kernel(
    const u16b* __restrict__ q, const u16b* __restrict__ k,
    const u16b* __restrict__ vT, const float* __restrict__ rb,
    const int* __restrict__ mask, u16b* __restrict__ ctx) {
  __shared__ float Sf[16 * 1024];  // 64 KB
  int tid  = threadIdx.x;
  int lane = tid & 31;
  int wid  = tid >> 5;  // 0..3
  int q0 = blockIdx.x * 16;
  int h  = blockIdx.y;
  int b  = blockIdx.z;
  size_t bh = (size_t)(b * 8 + h);
  const u16b* qb = q  + bh * 1024 * 64;
  const u16b* kb = k  + bh * 1024 * 64;
  const u16b* vb = vT + bh * 64 * 1024;
  const float* rbh = rb + (size_t)h * 1024 * 1024;

  // ---- Phase 1: S = Q@K^T + rel_bias, key-masked, into LDS ----
  FragAB fq0, fq1;
  load_fragA(fq0, qb + (size_t)q0 * 64 + 0,  64, lane);
  load_fragA(fq1, qb + (size_t)q0 * 64 + 32, 64, lane);
  for (int kt = wid * 16; kt < wid * 16 + 16; ++kt) {
    int key0 = kt * 16;
    FragC c;
#pragma unroll
    for (int j = 0; j < 8; ++j) c.f[j] = 0.f;
    FragAB fb;
    load_fragB(fb, kb + (size_t)key0 * 64 + 0, 64, lane);
    c.v = do_wmma(fq0, fb, c.v);
    load_fragB(fb, kb + (size_t)key0 * 64 + 32, 64, lane);
    c.v = do_wmma(fq1, fb, c.v);
    int key = key0 + (lane & 15);
    bool kvalid = (mask[b * 1024 + key] != 0);
#pragma unroll
    for (int j = 0; j < 8; ++j) {
      int ml = j + ((lane >> 4) << 3);
      float s = c.f[j] + rbh[(size_t)(q0 + ml) * 1024 + key];
      Sf[ml * 1024 + key] = kvalid ? s : -10000.0f;
    }
  }
  __syncthreads();

  // ---- Phase 2: softmax, 8 lanes per row, shuffle reductions ----
  {
    int row = tid >> 3;
    int sub = tid & 7;
    float* Srow = &Sf[row * 1024 + sub * 128];
    float mx = -3.0e38f;
    for (int i = 0; i < 128; ++i) mx = fmaxf(mx, Srow[i]);
#pragma unroll
    for (int o = 1; o < 8; o <<= 1) mx = fmaxf(mx, __shfl_xor(mx, o, 8));
    float sum = 0.f;
    for (int i = 0; i < 128; ++i) {
      float e = __expf(Srow[i] - mx);
      Srow[i] = e;
      sum += e;
    }
#pragma unroll
    for (int o = 1; o < 8; o <<= 1) sum += __shfl_xor(sum, o, 8);
    float inv = 1.0f / sum;
    for (int i = 0; i < 128; ++i) Srow[i] *= inv;
  }
  __syncthreads();

  // ---- Phase 3: ctx = P @ V, wave handles 16 of the 64 head dims ----
  int d0 = wid * 16;
  FragC c;
#pragma unroll
  for (int j = 0; j < 8; ++j) c.f[j] = 0.f;
  for (int k0 = 0; k0 < 1024; k0 += 32) {
    FragAB fp;
    {
      int mq = lane & 15;
      int kk = (lane >> 4) << 3;
      const float* p0 = &Sf[mq * 1024 + k0 + kk];
#pragma unroll
      for (int j = 0; j < 8; ++j) fp.s[j] = f2bf(p0[j]);
#pragma unroll
      for (int j = 0; j < 8; ++j) fp.s[8 + j] = f2bf(p0[16 + j]);
    }
    FragAB fv;
    load_fragB(fv, vb + (size_t)d0 * 1024 + k0, 1024, lane);
    c.v = do_wmma(fp, fv, c.v);
  }
#pragma unroll
  for (int j = 0; j < 8; ++j) {
    int ml = j + ((lane >> 4) << 3);
    int qi = q0 + ml;
    int d  = d0 + (lane & 15);
    float tf = (mask[b * 1024 + qi] != 0) ? 1.0f : 0.0f;
    ctx[((size_t)b * 1024 + qi) * 512 + h * 64 + d] = f2bf(c.f[j] * tf);
  }
}

// --------------------------- host launcher ---------------------------
extern "C" void kernel_launch(void* const* d_in, const int* in_sizes, int n_in,
                              void* d_out, int out_size, void* d_ws, size_t ws_size,
                              hipStream_t stream) {
  (void)in_sizes; (void)n_in; (void)out_size; (void)ws_size;
  const float* tokens   = (const float*)d_in[0];
  const int*   tmask    = (const int*)d_in[1];
  const float* rel_bias = (const float*)d_in[2];
  const float* ln1_s    = (const float*)d_in[3];
  const float* ln1_b    = (const float*)d_in[4];
  const float* qkv_w    = (const float*)d_in[5];
  const float* qkv_b    = (const float*)d_in[6];
  const float* out_w    = (const float*)d_in[7];
  const float* out_b    = (const float*)d_in[8];
  const float* ln2_s    = (const float*)d_in[9];
  const float* ln2_b    = (const float*)d_in[10];
  const float* fc1_w    = (const float*)d_in[11];
  const float* fc1_b    = (const float*)d_in[12];
  const float* fc2_w    = (const float*)d_in[13];
  const float* fc2_b    = (const float*)d_in[14];

  const int B = 8, L = 1024, C = 512, H = 8, FF = 2048;
  const int M = B * L;

  char* ws = (char*)d_ws;
  size_t off = 0;
  auto carve = [&](size_t bytes) -> void* {
    void* p = ws + off;
    off = (off + bytes + 255) & ~(size_t)255;
    return p;
  };
  u16b*  wt_qkv = (u16b*)carve((size_t)4 * 1536 * 512 * 2);
  u16b*  wt_out = (u16b*)carve((size_t)4 * 512 * 512 * 2);
  u16b*  wt_fc1 = (u16b*)carve((size_t)4 * 2048 * 512 * 2);
  u16b*  wt_fc2 = (u16b*)carve((size_t)4 * 512 * 2048 * 2);
  float* xbuf   = (float*)carve((size_t)M * C * 4);
  u16b*  h_bf   = (u16b*)carve((size_t)M * C * 2);
  u16b*  qb     = (u16b*)carve((size_t)B * H * L * 64 * 2);
  u16b*  kb     = (u16b*)carve((size_t)B * H * L * 64 * 2);
  u16b*  vtb    = (u16b*)carve((size_t)B * H * 64 * L * 2);
  u16b*  ctxb   = (u16b*)carve((size_t)M * C * 2);
  u16b*  h2_bf  = (u16b*)carve((size_t)M * C * 2);
  u16b*  ff_bf  = (u16b*)carve((size_t)M * FF * 2);

  dim3 blk(256);
  for (int i = 0; i < 4; ++i) {
    convT_kernel<<<(512 * 1536 + 255) / 256, blk, 0, stream>>>(
        qkv_w + (size_t)i * 512 * 1536, wt_qkv + (size_t)i * 1536 * 512, 512, 1536);
    convT_kernel<<<(512 * 512 + 255) / 256, blk, 0, stream>>>(
        out_w + (size_t)i * 512 * 512, wt_out + (size_t)i * 512 * 512, 512, 512);
    convT_kernel<<<(512 * 2048 + 255) / 256, blk, 0, stream>>>(
        fc1_w + (size_t)i * 512 * 2048, wt_fc1 + (size_t)i * 2048 * 512, 512, 2048);
    convT_kernel<<<(2048 * 512 + 255) / 256, blk, 0, stream>>>(
        fc2_w + (size_t)i * 2048 * 512, wt_fc2 + (size_t)i * 512 * 2048, 2048, 512);
  }
  copy_f32_kernel<<<(M * C + 255) / 256, blk, 0, stream>>>(tokens, xbuf, M * C);

  const float qscale = 0.125f;  // 1/sqrt(64)

  for (int i = 0; i < 4; ++i) {
    ln_kernel<<<M / 8, blk, 0, stream>>>(xbuf, ln1_s + i * C, ln1_b + i * C, h_bf);

    gemm_bf16<1><<<dim3(1536 / 128, M / 128), blk, 0, stream>>>(
        h_bf, wt_qkv + (size_t)i * 1536 * 512, qkv_b + (size_t)i * 1536,
        M, 1536, 512, nullptr, nullptr, nullptr, nullptr, qb, kb, vtb, qscale);

    attn_kernel<<<dim3(L / 16, H, B), dim3(128), 0, stream>>>(
        qb, kb, vtb, rel_bias + (size_t)i * H * L * L, tmask, ctxb);

    gemm_bf16<2><<<dim3(512 / 128, M / 128), blk, 0, stream>>>(
        ctxb, wt_out + (size_t)i * 512 * 512, out_b + (size_t)i * 512,
        M, 512, 512, nullptr, xbuf, xbuf, nullptr, nullptr, nullptr, nullptr, 1.f);

    ln_kernel<<<M / 8, blk, 0, stream>>>(xbuf, ln2_s + i * C, ln2_b + i * C, h2_bf);

    gemm_bf16<3><<<dim3(2048 / 128, M / 128), blk, 0, stream>>>(
        h2_bf, wt_fc1 + (size_t)i * 2048 * 512, fc1_b + (size_t)i * 2048,
        M, 2048, 512, ff_bf, nullptr, nullptr, nullptr, nullptr, nullptr, nullptr, 1.f);

    float* dst = (i == 3) ? (float*)d_out : xbuf;
    gemm_bf16<4><<<dim3(512 / 128, M / 128), blk, 0, stream>>>(
        ff_bf, wt_fc2 + (size_t)i * 512 * 2048, fc2_b + (size_t)i * 512,
        M, 512, 2048, nullptr, dst, xbuf, tmask, nullptr, nullptr, nullptr, 1.f);
  }
}
